// CausalConv_67276367725084
// MI455X (gfx1250) — matrix operand order repaired and checked by reading, not compiled, not run on hardware
//
#include <hip/hip_runtime.h>

// ---------------- CDNA5 types ----------------
typedef __attribute__((ext_vector_type(16))) __bf16        v16bf;
typedef __attribute__((ext_vector_type(8)))  float         v8f;
typedef __attribute__((ext_vector_type(4)))  unsigned int  v4u;

#define B_   32
#define T_   8192
#define N_   128
#define KT   3
#define TM   4                  // 16-row time tiles per workgroup
#define ROWS (TM * 16 + 4)      // 68 rows staged (4 rows of causal history)
#define LDSW 136                // padded row stride (bf16): 68 dwords -> conflict-free

// =====================================================================
// Prep kernel: pack weight[o,i,j] (f32) into bf16 hi/lo B-fragments in
// a contiguous-k 32x16 bf16 B layout (A uses the same k order):
//   lane l: column n = l&15, k-base = 16*(l>>4); q=0..15 -> k = kbase+q
// fragment f = ((j*4 + kb)*8 + ntile); element i = kb*32 + k, o = ntile*16 + n
// hi plane at ws[f*512 + l*16 + q], lo plane offset by 96*512.
// =====================================================================
__global__ void pack_weights_kernel(const float* __restrict__ w,
                                    unsigned short* __restrict__ ws) {
    int gid = blockIdx.x * blockDim.x + threadIdx.x;
    if (gid >= 96 * 512) return;
    int f     = gid >> 9;
    int rem   = gid & 511;
    int l     = rem >> 4;
    int q     = rem & 15;
    int ntile = f & 7;
    int kb    = (f >> 3) & 3;
    int j     = f >> 5;
    int n     = l & 15;
    int kbase = (l >> 4) << 4;
    int i     = kb * 32 + kbase + q;
    int o     = ntile * 16 + n;

    float v  = w[(o * N_ + i) * KT + j];
    __bf16 hi = (__bf16)v;
    __bf16 lo = (__bf16)(v - (float)hi);
    ws[gid]            = __builtin_bit_cast(unsigned short, hi);
    ws[96 * 512 + gid] = __builtin_bit_cast(unsigned short, lo);
}

// =====================================================================
// Main kernel: out[b,t,o] = bias[o] + sum_j sum_i x[b,t-2j,i]*W[o,i,j]
// grid = (T/64, B), block = 256 (8 waves); wave w -> out chans [16w,16w+16)
// =====================================================================
__global__ __launch_bounds__(256) void causal_conv_wmma_kernel(
        const float* __restrict__ x,
        const unsigned short* __restrict__ wfrag_raw,
        const float* __restrict__ bias,
        float* __restrict__ out) {
    __shared__ __bf16 xhi[ROWS * LDSW];
    __shared__ __bf16 xlo[ROWS * LDSW];

    const int tid = threadIdx.x;
    const int b   = blockIdx.y;
    const int t0  = blockIdx.x * (TM * 16);

    // ---- cooperative stage: split x tile into bf16 hi/lo planes ----
    const float* xb = x + (size_t)b * T_ * N_;
    for (int idx = tid; idx < ROWS * 32; idx += 256) {
        int row = idx >> 5;          // 0..67
        int c4  = idx & 31;          // float4 column
        int t   = t0 - 4 + row;      // causal history rows may be < 0
        float4 v = make_float4(0.f, 0.f, 0.f, 0.f);
        if (t >= 0) v = *(const float4*)(xb + (size_t)t * N_ + c4 * 4);
        float vv[4] = {v.x, v.y, v.z, v.w};
        int base = row * LDSW + c4 * 4;
#pragma unroll
        for (int e = 0; e < 4; ++e) {
            __bf16 h = (__bf16)vv[e];
            xhi[base + e] = h;
            xlo[base + e] = (__bf16)(vv[e] - (float)h);
        }
    }
    __syncthreads();

    const int l    = tid & 31;
    const int wave = tid >> 5;       // ntile
    const int n    = l & 15;         // also A-matrix row m
    const int sub  = l >> 4;
    const int o0   = wave * 16;
    const __bf16* wfrag = (const __bf16*)wfrag_raw;

    // C/D tiles: N = lane&15 (out chan), M = r + 8*sub (time)
    float bv = bias[o0 + n];
    v8f c[TM];
#pragma unroll
    for (int tm = 0; tm < TM; ++tm)
#pragma unroll
        for (int r = 0; r < 8; ++r) c[tm][r] = bv;

#pragma unroll
    for (int j = 0; j < KT; ++j) {
#pragma unroll
        for (int kb = 0; kb < 4; ++kb) {
            // ---- B fragments (pre-packed, coalesced b128 loads) ----
            int f = (j * 4 + kb) * 8 + wave;
            const v4u* bph = (const v4u*)(wfrag + f * 512 + l * 16);
            const v4u* bpl = (const v4u*)(wfrag + 96 * 512 + f * 512 + l * 16);
            union { v4u u[2]; v16bf v; } bh, bl;
            bh.u[0] = bph[0]; bh.u[1] = bph[1];
            bl.u[0] = bpl[0]; bl.u[1] = bpl[1];

#pragma unroll
            for (int tm = 0; tm < TM; ++tm) {
                // A layout: lane m = l&15; k-run [16*sub,+16)
                int row  = tm * 16 + n + 4 - 2 * j;   // shifted time row in LDS
                int eoff = row * LDSW + kb * 32 + 16 * sub;
                union { v4u u[2]; v16bf v; } ah, al;
                ah.u[0] = *(const v4u*)(xhi + eoff);
                ah.u[1] = *(const v4u*)(xhi + eoff + 8);
                al.u[0] = *(const v4u*)(xlo + eoff);
                al.u[1] = *(const v4u*)(xlo + eoff + 8);

                // 3-term bf16 split ~= fp32 GEMM, f32 accumulate
                c[tm] = __builtin_amdgcn_wmma_f32_16x16x32_bf16(
                            false, al.v, false, bh.v, (short)0, c[tm], false, false);
                c[tm] = __builtin_amdgcn_wmma_f32_16x16x32_bf16(
                            false, ah.v, false, bl.v, (short)0, c[tm], false, false);
                c[tm] = __builtin_amdgcn_wmma_f32_16x16x32_bf16(
                            false, ah.v, false, bh.v, (short)0, c[tm], false, false);
            }
        }
    }

    // ---- store: out[(b*T + t0 + tm*16 + M)*128 + o0 + N] ----
    float* ob = out + ((size_t)b * T_ + t0) * N_ + o0 + n;
#pragma unroll
    for (int tm = 0; tm < TM; ++tm)
#pragma unroll
        for (int r = 0; r < 8; ++r) {
            int m = r + 8 * sub;
            ob[(size_t)(tm * 16 + m) * N_] = c[tm][r];
        }
}

extern "C" void kernel_launch(void* const* d_in, const int* in_sizes, int n_in,
                              void* d_out, int out_size, void* d_ws, size_t ws_size,
                              hipStream_t stream) {
    const float* x    = (const float*)d_in[0];
    const float* w    = (const float*)d_in[1];
    const float* bias = (const float*)d_in[2];
    float* out        = (float*)d_out;
    unsigned short* ws = (unsigned short*)d_ws;   // 2 * 96 * 512 bf16 = 192 KB

    pack_weights_kernel<<<(96 * 512 + 255) / 256, 256, 0, stream>>>(w, ws);

    dim3 grid(T_ / (TM * 16), B_);
    causal_conv_wmma_kernel<<<grid, 256, 0, stream>>>(x, ws, bias, out);
}
